// LinearAttention_37915971289751
// MI455X (gfx1250) — matrix operand order repaired and checked by reading, not compiled, not run on hardware
//
#include <hip/hip_runtime.h>
#include <hip/hip_bf16.h>

// ---------- types ----------
typedef __attribute__((ext_vector_type(16))) __bf16 v16bf;
typedef __attribute__((ext_vector_type(8)))  float  v8f;
typedef __attribute__((ext_vector_type(4)))  float  v4f;
typedef unsigned short u16;
typedef __attribute__((ext_vector_type(8))) unsigned short u16x8;
typedef __attribute__((ext_vector_type(4))) unsigned short u16x4;

// ---------- bf16 helpers (bit-level, RNE) ----------
__device__ __forceinline__ u16 f2bf(float f) {
    union { float f; unsigned u; } v; v.f = f;
    unsigned u = v.u;
    u += 0x7FFFu + ((u >> 16) & 1u);
    return (u16)(u >> 16);
}
__device__ __forceinline__ float bf2f(u16 h) {
    union { unsigned u; float f; } v; v.u = ((unsigned)h) << 16;
    return v.f;
}

// ---------- WMMA fragment builders: 2x ds_load_b128 each ----------
// A 16x32 (row-major tile [m][k], ld mult of 8):
//   lane: m = L&15, half = L>>4; elems = K[8h..8h+7] ++ K[16+8h..16+8h+7]
__device__ __forceinline__ v16bf load_a_frag(const u16* p, int ld, int koff) {
    const int lane = threadIdx.x & 31, m = lane & 15, half = lane >> 4;
    const u16* q = p + m * ld + koff + (half << 3);
    union { v16bf v; u16x8 h[2]; } u;
    u.h[0] = *(const u16x8*)q;
    u.h[1] = *(const u16x8*)(q + 16);
    return u.v;
}
// B 32x16 from a TRANSPOSED tile [n][k] (ld mult of 8):
//   lane: n = L&15, half = L>>4; elems = K[8h..8h+7] ++ K[16+8h..16+8h+7]
__device__ __forceinline__ v16bf load_b_fragT(const u16* p, int ld, int koff) {
    const int lane = threadIdx.x & 31, n = lane & 15, half = lane >> 4;
    const u16* q = p + n * ld + koff + (half << 3);
    union { v16bf v; u16x8 h[2]; } u;
    u.h[0] = *(const u16x8*)q;
    u.h[1] = *(const u16x8*)(q + 16);
    return u.v;
}
__device__ __forceinline__ v8f wmma_bf16(v16bf a, v16bf b, v8f c) {
    return __builtin_amdgcn_wmma_f32_16x16x32_bf16(
        false, a, false, b, (short)0, c, false, false);
}

// ============================================================================
// Kernel 1/4: 512-wide GEMM with fused epilogues, software-pipelined:
// register-staged global loads + double-buffered LDS, one barrier per K-step.
//   MODE 0: out = A@W + b                      (bf16 out) -> V
//   MODE 1: out = softmax64(A@W + b)           (bf16 out) -> Q
//   MODE 2: MODE 1 + atomic per-(b,col) colsum (bf16 out) -> K + ksum
//   MODE 3: out = A@W + b                      (f32  out) -> final proj
// ============================================================================
template <int MODE, bool ABF16>
__global__ __launch_bounds__(128) void gemm512(
    const void* __restrict__ Ap, const float* __restrict__ W,
    const float* __restrict__ bias, void* __restrict__ Outp,
    float* __restrict__ ksum, int rowsPerBatch)
{
    constexpr int Cd = 512, ALD = 40, BLD = 40, KITERS = Cd / 32;
    __shared__ __attribute__((aligned(16))) u16 As[2][64 * ALD];  // [m][k]
    __shared__ __attribute__((aligned(16))) u16 Bt[2][64 * BLD];  // [n][k] transposed
    const int col0 = blockIdx.x * 64, row0 = blockIdx.y * 64;
    const int t = threadIdx.x, wave = t >> 5, lane = t & 31;
    const float* Af = (const float*)Ap;
    const u16*   Ab = (const u16*)Ap;

    v4f   aF[4];            // staged A tile (fp32 path)
    u16x4 aB[4];            // staged A tile (bf16 path)
    float bF[4][4];         // staged B tile (4 K-rows per slot)

    auto issue_loads = [&](int k0) {
#pragma unroll
        for (int i = 0; i < 4; ++i) {                 // A tile 64x32, b128 loads
            const int idx = t + i * 128, r = idx >> 3, c4 = (idx & 7) << 2;
            if constexpr (ABF16) aB[i] = *(const u16x4*)&Ab[(size_t)(row0 + r) * Cd + k0 + c4];
            else                 aF[i] = *(const v4f*)&Af[(size_t)(row0 + r) * Cd + k0 + c4];
        }
#pragma unroll
        for (int i = 0; i < 4; ++i) {                 // B tile 32x64, coalesced rows
            const int idx = t + i * 128, kg = (idx >> 6) << 2, c = idx & 63;
#pragma unroll
            for (int j = 0; j < 4; ++j)
                bF[i][j] = W[(size_t)(k0 + kg + j) * Cd + col0 + c];
        }
    };
    auto store_lds = [&](int buf) {
#pragma unroll
        for (int i = 0; i < 4; ++i) {
            const int idx = t + i * 128, r = idx >> 3, c4 = (idx & 7) << 2;
            u16x4 pk;
            if constexpr (ABF16) pk = aB[i];
            else {
#pragma unroll
                for (int j = 0; j < 4; ++j) pk[j] = f2bf(aF[i][j]);
            }
            *(u16x4*)&As[buf][r * ALD + c4] = pk;
        }
#pragma unroll
        for (int i = 0; i < 4; ++i) {                 // transposed [n][k] store
            const int idx = t + i * 128, kg = (idx >> 6) << 2, c = idx & 63;
            u16x4 pk;
#pragma unroll
            for (int j = 0; j < 4; ++j) pk[j] = f2bf(bF[i][j]);
            *(u16x4*)&Bt[buf][c * BLD + kg] = pk;
        }
    };

    v8f acc[4] = {};
    issue_loads(0);
    store_lds(0);
    __syncthreads();
    for (int it = 0; it < KITERS; ++it) {
        if (it + 1 < KITERS) issue_loads((it + 1) * 32);   // next tile in flight
        const u16* As_ = As[it & 1];
        const u16* Bt_ = Bt[it & 1];
        const v16bf a = load_a_frag(As_ + wave * 16 * ALD, ALD, 0);
#pragma unroll
        for (int c = 0; c < 4; ++c) {
            const v16bf b = load_b_fragT(Bt_ + (c * 16) * BLD, BLD, 0);
            acc[c] = wmma_bf16(a, b, acc[c]);
        }
        if (it + 1 < KITERS) store_lds((it + 1) & 1);
        __syncthreads();
    }

    const int n = lane & 15, half = lane >> 4;
#pragma unroll
    for (int c = 0; c < 4; ++c) {                    // fused bias
        const float bv = bias[col0 + c * 16 + n];
#pragma unroll
        for (int r = 0; r < 8; ++r) acc[c][r] += bv;
    }

    if constexpr (MODE == 1 || MODE == 2) {          // fused softmax over 64 cols
#pragma unroll
        for (int r = 0; r < 8; ++r) {
            float mx = acc[0][r];
#pragma unroll
            for (int c = 1; c < 4; ++c) mx = fmaxf(mx, acc[c][r]);
#pragma unroll
            for (int off = 1; off < 16; off <<= 1) mx = fmaxf(mx, __shfl_xor(mx, off, 32));
            float s = 0.f;
#pragma unroll
            for (int c = 0; c < 4; ++c) { const float e = __expf(acc[c][r] - mx); acc[c][r] = e; s += e; }
#pragma unroll
            for (int off = 1; off < 16; off <<= 1) s += __shfl_xor(s, off, 32);
            const float inv = 1.0f / s;
#pragma unroll
            for (int c = 0; c < 4; ++c) acc[c][r] *= inv;
        }
    }
    if constexpr (MODE == 2) {                       // fused ksum (column sums)
        const int b = row0 / rowsPerBatch;
#pragma unroll
        for (int c = 0; c < 4; ++c) {
            float cs = 0.f;
#pragma unroll
            for (int r = 0; r < 8; ++r) cs += acc[c][r];
            cs += __shfl_xor(cs, 16, 32);
            if (lane < 16) atomicAdd(&ksum[b * Cd + col0 + c * 16 + n], cs);
        }
    }

    if constexpr (MODE == 3) {
        float* O = (float*)Outp;
#pragma unroll
        for (int c = 0; c < 4; ++c)
#pragma unroll
            for (int r = 0; r < 8; ++r)
                O[(size_t)(row0 + wave * 16 + r + half * 8) * Cd + col0 + c * 16 + n] = acc[c][r];
    } else {
        u16* O = (u16*)Outp;
#pragma unroll
        for (int c = 0; c < 4; ++c)
#pragma unroll
            for (int r = 0; r < 8; ++r)
                O[(size_t)(row0 + wave * 16 + r + half * 8) * Cd + col0 + c * 16 + n] = f2bf(acc[c][r]);
    }
}

// ============================================================================
// Kernel 2: kv[b,h] = k[b,:,h]^T @ v[b,:,h]  (64x64, reduce over M in chunks)
// Tiles stored transposed ([d][m] / [e][m]); same register+LDS double-buffer
// pipeline as the GEMM.
// ============================================================================
__global__ __launch_bounds__(128) void kv_kernel(
    const u16* __restrict__ Kb, const u16* __restrict__ Vb,
    float* __restrict__ KV, int M, int mchunk)
{
    constexpr int LD = 40;
    __shared__ __attribute__((aligned(16))) u16 kt[2][64 * LD];  // [d][m]
    __shared__ __attribute__((aligned(16))) u16 vt[2][64 * LD];  // [e][m]
    const int m0 = blockIdx.x * mchunk, h = blockIdx.y, b = blockIdx.z;
    const int t = threadIdx.x, wave = t >> 5, lane = t & 31;

    u16x4 ka[4], va[4];
    auto issue_loads = [&](int mt) {
#pragma unroll
        for (int i = 0; i < 4; ++i) {
            const int idx = t + i * 128, mg = (idx >> 6) << 2, d = idx & 63;
            const size_t g = ((size_t)(b * M + mt + mg)) * 512 + h * 64 + d;
#pragma unroll
            for (int j = 0; j < 4; ++j) {
                ka[i][j] = Kb[g + (size_t)j * 512];
                va[i][j] = Vb[g + (size_t)j * 512];
            }
        }
    };
    auto store_lds = [&](int buf) {
#pragma unroll
        for (int i = 0; i < 4; ++i) {
            const int idx = t + i * 128, mg = (idx >> 6) << 2, d = idx & 63;
            *(u16x4*)&kt[buf][d * LD + mg] = ka[i];
            *(u16x4*)&vt[buf][d * LD + mg] = va[i];
        }
    };

    const int iters = mchunk / 32;
    v8f acc[4] = {};
    issue_loads(m0);
    store_lds(0);
    __syncthreads();
    for (int it = 0; it < iters; ++it) {
        if (it + 1 < iters) issue_loads(m0 + (it + 1) * 32);
        const u16* kt_ = kt[it & 1];
        const u16* vt_ = vt[it & 1];
        const v16bf a = load_a_frag(kt_ + wave * 16 * LD, LD, 0);  // A = k^T
#pragma unroll
        for (int c = 0; c < 4; ++c) {
            const v16bf bb = load_b_fragT(vt_ + (c * 16) * LD, LD, 0);
            acc[c] = wmma_bf16(a, bb, acc[c]);
        }
        if (it + 1 < iters) store_lds((it + 1) & 1);
        __syncthreads();
    }
    const int n = lane & 15, half = lane >> 4;
#pragma unroll
    for (int c = 0; c < 4; ++c)
#pragma unroll
        for (int r = 0; r < 8; ++r) {
            const int d = wave * 16 + r + half * 8, e = c * 16 + n;
            atomicAdd(&KV[(size_t)((b * 8 + h) * 64 + d) * 64 + e], acc[c][r]);
        }
}

// ============================================================================
// Kernel 3: attn = q @ kv / denom + q, denom = max(sum_d q*ksum, 1e-6)
// ============================================================================
__global__ __launch_bounds__(128) void attn_kernel(
    const u16* __restrict__ Qb, const float* __restrict__ KVb,
    const float* __restrict__ KS, u16* __restrict__ ATT, int N)
{
    constexpr int LD = 72;
    __shared__ __attribute__((aligned(16))) u16 qs[64 * LD];    // [row][d]
    __shared__ __attribute__((aligned(16))) u16 kvt[64 * LD];   // [e][d] transposed
    __shared__ float dsum[128];
    __shared__ float denom[64];
    const int h = blockIdx.x, row0 = blockIdx.y * 64;
    const int b = row0 / N;
    const int t = threadIdx.x, wave = t >> 5, lane = t & 31;
#pragma unroll
    for (int i = 0; i < 8; ++i) {                    // q tile 64x64 (b64 copies)
        const int idx = t + i * 128, r = idx >> 4, c4 = (idx & 15) << 2;
        *(u16x4*)&qs[r * LD + c4] = *(const u16x4*)&Qb[(size_t)(row0 + r) * 512 + h * 64 + c4];
    }
#pragma unroll
    for (int i = 0; i < 8; ++i) {                    // kv 64x64 -> transposed [e][d]
        const int idx = t + i * 128, dg = (idx >> 6) << 2, e = idx & 63;
        u16x4 pk;
#pragma unroll
        for (int j = 0; j < 4; ++j)
            pk[j] = f2bf(KVb[(size_t)((b * 8 + h) * 64 + dg + j) * 64 + e]);
        *(u16x4*)&kvt[e * LD + dg] = pk;
    }
    __syncthreads();
    {   // per-row denominator: 2 threads per row, 32 elems each
        const int row = t >> 1, part = t & 1;
        float s = 0.f;
#pragma unroll
        for (int dg = 0; dg < 8; ++dg) {
            const u16x4 qv = *(const u16x4*)&qs[row * LD + part * 32 + dg * 4];
#pragma unroll
            for (int j = 0; j < 4; ++j)
                s += bf2f(qv[j]) * KS[b * 512 + h * 64 + part * 32 + dg * 4 + j];
        }
        dsum[t] = s;
    }
    __syncthreads();
    if (t < 64) denom[t] = fmaxf(dsum[2 * t] + dsum[2 * t + 1], 1e-6f);
    __syncthreads();

    v8f acc[4] = {};
#pragma unroll
    for (int kt2 = 0; kt2 < 2; ++kt2) {              // K = 64, two 32-steps
        const v16bf a = load_a_frag(qs + wave * 16 * LD, LD, kt2 * 32);
#pragma unroll
        for (int c = 0; c < 4; ++c) {
            const v16bf bb = load_b_fragT(kvt + (c * 16) * LD, LD, kt2 * 32);
            acc[c] = wmma_bf16(a, bb, acc[c]);
        }
    }
    const int n = lane & 15, half = lane >> 4;
#pragma unroll
    for (int c = 0; c < 4; ++c)
#pragma unroll
        for (int r = 0; r < 8; ++r) {
            const int rl = wave * 16 + r + half * 8;
            const float qv = bf2f(qs[rl * LD + c * 16 + n]);
            const float val = acc[c][r] / denom[rl] + qv;
            ATT[(size_t)(row0 + rl) * 512 + h * 64 + c * 16 + n] = f2bf(val);
        }
}

// ============================================================================
extern "C" void kernel_launch(void* const* d_in, const int* in_sizes, int n_in,
                              void* d_out, int out_size, void* d_ws, size_t ws_size,
                              hipStream_t stream) {
    (void)in_sizes; (void)n_in; (void)out_size; (void)ws_size;
    const float* x  = (const float*)d_in[0];
    const float* y  = (const float*)d_in[1];
    const float* Wq = (const float*)d_in[2];  const float* bq = (const float*)d_in[3];
    const float* Wk = (const float*)d_in[4];  const float* bk = (const float*)d_in[5];
    const float* Wv = (const float*)d_in[6];  const float* bv = (const float*)d_in[7];
    const float* Wp = (const float*)d_in[8];  const float* bp = (const float*)d_in[9];

    constexpr int B = 4, N = 8192, M = 8192, Cd = 512, H = 8;
    const size_t rowsX = (size_t)B * N, rowsY = (size_t)B * M;

    char* ws = (char*)d_ws;
    size_t off = 0;
    u16* Qb  = (u16*)(ws + off); off += rowsX * Cd * sizeof(u16);
    u16* Kb  = (u16*)(ws + off); off += rowsY * Cd * sizeof(u16);
    u16* Vb  = (u16*)(ws + off); off += rowsY * Cd * sizeof(u16);
    u16* ATT = (u16*)(ws + off); off += rowsX * Cd * sizeof(u16);
    float* KV = (float*)(ws + off); off += (size_t)B * H * 64 * 64 * sizeof(float);
    float* KS = (float*)(ws + off); off += (size_t)B * Cd * sizeof(float);

    hipMemsetAsync(KV, 0, (size_t)B * H * 64 * 64 * sizeof(float), stream);
    hipMemsetAsync(KS, 0, (size_t)B * Cd * sizeof(float), stream);

    const dim3 blk(128);
    gemm512<1, false><<<dim3(Cd / 64, rowsX / 64), blk, 0, stream>>>(x, Wq, bq, Qb, nullptr, N);
    gemm512<2, false><<<dim3(Cd / 64, rowsY / 64), blk, 0, stream>>>(y, Wk, bk, Kb, KS, M);
    gemm512<0, false><<<dim3(Cd / 64, rowsY / 64), blk, 0, stream>>>(y, Wv, bv, Vb, nullptr, M);
    kv_kernel<<<dim3(M / 1024, H, B), blk, 0, stream>>>(Kb, Vb, KV, M, 1024);
    attn_kernel<<<dim3(H, rowsX / 64), blk, 0, stream>>>(Qb, KV, KS, ATT, N);
    gemm512<3, true><<<dim3(Cd / 64, rowsX / 64), blk, 0, stream>>>(ATT, Wp, bp, (float*)d_out, nullptr, N);
}
